// ParallelCNN_76312978915550
// MI455X (gfx1250) — matrix operand order repaired and checked by reading, not compiled, not run on hardware
//
#include <hip/hip_runtime.h>
#include <hip/hip_bf16.h>
#include <math.h>

typedef __attribute__((ext_vector_type(16))) _Float16 v16h;
typedef __attribute__((ext_vector_type(8)))  float    v8f;

#define IMG_H 28
#define IMG_W 28
#define PPI   784                       // pixels per image
#define NUM_IMG 16384
#define NPIX  (NUM_IMG * PPI)           // 12,845,056
#define NTILES (NPIX / 16)              // 802,816 exact (784 = 49*16)
#define TPI   49                        // tiles per image
#define IMGS_PER_WAVE 2                 // 8192 waves * 2 = 16384 images

#define LOG2E 1.4426950408889634f
#define LN2   0.6931471805599453f

// parity class: (h even,w even)->0, (odd,odd)->1, (even,odd)->2, (odd,even)->3
__device__ __forceinline__ int pclass(int h, int w) {
    return ((h & 1) == 0) ? ((w & 1) ? 2 : 0) : ((w & 1) ? 1 : 3);
}

// stable softplus via raw hw transcendentals; log2 argument is in [1,2] so
// no denorm/inf fixups are needed: max(x,0) + ln2*log2(1 + exp2(-|x|*log2e))
__device__ __forceinline__ float softplus_fast(float x) {
    const float e = __builtin_amdgcn_exp2f(-fabsf(x) * LOG2E);
    return fmaxf(x, 0.0f) + LN2 * __builtin_amdgcn_logf(1.0f + e);
}

__global__ void pcnn_zero_out(float* out) { out[0] = 0.0f; }

// posLogit[i] = bias[cls(i)] + sum_t position[q_t] * w[cls(i)][ch1][t]
// (position-channel conv + bias depends only on pixel-in-image, not batch)
__global__ __launch_bounds__(256) void pcnn_poslogit(
    const float* __restrict__ position, const float* __restrict__ weights,
    const float* __restrict__ biases, float* __restrict__ posLogit)
{
    const int i = blockIdx.x * blockDim.x + threadIdx.x;
    if (i >= PPI) return;
    const int h = i / IMG_W;
    const int w = i - h * IMG_W;
    const int c = pclass(h, w);
    float acc = biases[c];
#pragma unroll
    for (int t = 0; t < 9; ++t) {
        const int dh = t / 3 - 1, dw = t % 3 - 1;
        const int hh = h + dh, ww = w + dw;
        if (hh >= 0 && hh < IMG_H && ww >= 0 && ww < IMG_W)
            acc += position[hh * IMG_W + ww] * weights[c * 18 + 9 + t];
    }
    posLogit[i] = acc;
}

__global__ __launch_bounds__(256) void pcnn_wmma_kernel(
    const float* __restrict__ samples,   // (B,1,28,28)
    const float* __restrict__ weights,   // (4,1,2,3,3) flat = 72
    const float* __restrict__ posLogit,  // 784 precomputed
    float* __restrict__ out)
{
    const int tid    = threadIdx.x;
    const int lane   = tid & 31;
    const int gwave  = (blockIdx.x * blockDim.x + tid) >> 5;
    const int n      = lane & 15;        // row-in-tile / B-column index
    const bool hiH   = lane >= 16;
    const _Float16 hz = (_Float16)0.0f;

    // ---- B (32x16 f16, KxN): col n<4 = branch-n channel-0 taps, split into
    // f16 hi part at K0..8 (elems 0-7) and lo part at K16..24 (elems 8-15).
    // A duplicates the exact 0/1 taps at both K ranges -> ~f32-precise dot.
    v16h bmat;
#pragma unroll
    for (int e = 0; e < 16; ++e) {
        const int tp = hiH ? ((e & 7) ? 9 : 8) : (e & 7);
        const float wv  = (n < 4 && tp < 9) ? weights[n * 18 + tp] : 0.0f;
        const _Float16 whi = (_Float16)wv;
        const float wlo = wv - (float)whi;
        bmat[e] = (e >= 8) ? (_Float16)wlo : whi;
    }

    float acc = 0.0f;

    // contiguous ownership: wave g owns images 2g, 2g+1 (98 = 2*49 tiles)
    const int b0 = gwave * IMGS_PER_WAVE;
#pragma unroll 1
    for (int img = 0; img < IMGS_PER_WAVE; ++img) {
        const int ibase = (b0 + img) * PPI;   // flat image base
        int h0 = 0, w0 = 0;                   // coords of tile's first pixel
#pragma unroll 1
        for (int ti = 0; ti < TPI; ++ti) {
            const int t16   = ti * 16;
            const int pbase = ibase + t16;
            const int wrapAt = IMG_W - w0;    // lanes >= wrapAt are on h0+1

            // per-lane pixel (28 even -> w-parity ignores the wrap)
            const int wrapped = (n >= wrapAt) ? 1 : 0;
            const int h  = h0 + wrapped;
            const int w  = w0 + n - (wrapped ? IMG_W : 0);
            const int p  = pbase + n;
            const int hp = h & 1;
            const int wp = (w0 + n) & 1;

            // branch-free neighbor mask: corners iff hp, edges iff hp^wp
            const int mbits = (hp ? 0x145 : 0) | ((hp ^ wp) ? 0x0AA : 0);
            const bool okN = h > 0, okS = h < IMG_H - 1;
            const bool okW = w > 0, okE = w < IMG_W - 1;

            // gather 8 masked neighbor taps + raw center (center always masked)
            float st[9], sv4;
#pragma unroll
            for (int t = 0; t < 9; ++t) {
                if (t == 4) { sv4 = samples[p]; st[4] = 0.0f; continue; }
                const int dh = t / 3 - 1, dw = t % 3 - 1;
                const bool ok = (dh < 0 ? okN : dh > 0 ? okS : true) &&
                                (dw < 0 ? okW : dw > 0 ? okE : true);
                const int off = dh * IMG_W + dw;
                const float sv = samples[ok ? (p + off) : p];
                st[t] = (ok && ((mbits >> t) & 1)) ? sv : 0.0f;
            }

            // A (16x32 f16): lanes 0-15 elems = K0..7|K16..23,
            // lanes 16-31 elems = K8..15|K24..31; taps at K0..8 and K16..24
            v16h a;
            {
                const _Float16 s8h = (_Float16)st[8];
                a[0] = hiH ? s8h : (_Float16)st[0];
                a[1] = hiH ? hz  : (_Float16)st[1];
                a[2] = hiH ? hz  : (_Float16)st[2];
                a[3] = hiH ? hz  : (_Float16)st[3];
                a[4] = hz;
                a[5] = hiH ? hz  : (_Float16)st[5];
                a[6] = hiH ? hz  : (_Float16)st[6];
                a[7] = hiH ? hz  : (_Float16)st[7];
#pragma unroll
                for (int e = 0; e < 8; ++e) a[8 + e] = a[e];
            }

            v8f c = {};
            c = __builtin_amdgcn_wmma_f32_16x16x32_f16(
                    false, a, false, bmat, (short)0, c, false, false);

            // ---- epilogue: lane p(=n) fetches its own-class logit via
            // bpermute (D: lane c holds col c rows 0-7, lane c+16 rows 8-15)
            const int cidx = (hp << 1) | wp;
            const int cls  = (0x78 >> (cidx << 1)) & 3;   // {0,2,3,1} LUT
            const int srcLane = cls + ((n & 8) << 1);     // +16 for rows 8-15

            const float r0 = __shfl(c[0], srcLane, 32);
            const float r1 = __shfl(c[1], srcLane, 32);
            const float r2 = __shfl(c[2], srcLane, 32);
            const float r3 = __shfl(c[3], srcLane, 32);
            const float r4 = __shfl(c[4], srcLane, 32);
            const float r5 = __shfl(c[5], srcLane, 32);
            const float r6 = __shfl(c[6], srcLane, 32);
            const float r7 = __shfl(c[7], srcLane, 32);
            const int s3 = lane & 7;
            const float x0 = (s3 & 1) ? r1 : r0;
            const float x1 = (s3 & 1) ? r3 : r2;
            const float x2 = (s3 & 1) ? r5 : r4;
            const float x3 = (s3 & 1) ? r7 : r6;
            const float y0 = (s3 & 2) ? x1 : x0;
            const float y1 = (s3 & 2) ? x3 : x2;
            const float mm = (s3 & 4) ? y1 : y0;

            const float lg = mm + posLogit[t16 + n];
            const float sp = softplus_fast(lg);
            const float contrib = sv4 * lg - sp;   // sv4 == samples[pixel n]
            acc += (lane < 16) ? contrib : 0.0f;

            // speculative prefetch a few tiles ahead
            __builtin_prefetch(samples + p + 64, 0, 0);

            // advance wave-uniform tile coords (no divisions)
            w0 += 16;
            if (w0 >= IMG_W) { w0 -= IMG_W; h0 += 1; }
        }
    }

    // ---- block reduction, one atomic per block ---------------------------
    __shared__ float red[256];
    red[tid] = acc;
    __syncthreads();
#pragma unroll
    for (int s = 128; s > 0; s >>= 1) {
        if (tid < s) red[tid] += red[tid + s];
        __syncthreads();
    }
    if (tid == 0) atomicAdd(out, red[0]);
}

extern "C" void kernel_launch(void* const* d_in, const int* in_sizes, int n_in,
                              void* d_out, int out_size, void* d_ws, size_t ws_size,
                              hipStream_t stream) {
    const float* samples  = (const float*)d_in[0];
    const float* position = (const float*)d_in[1];
    const float* weights  = (const float*)d_in[2];
    const float* biases   = (const float*)d_in[3];
    float* out      = (float*)d_out;
    float* posLogit = (float*)d_ws;      // 784 floats of scratch

    pcnn_zero_out<<<1, 1, 0, stream>>>(out);
    pcnn_poslogit<<<(PPI + 255) / 256, 256, 0, stream>>>(position, weights,
                                                         biases, posLogit);
    // exactly 1024 blocks * 8 waves = 8192 waves; each owns 2 images
    pcnn_wmma_kernel<<<1024, 256, 0, stream>>>(samples, weights, posLogit, out);
}
